// MultiHeadAttention_88218628260740
// MI455X (gfx1250) — matrix operand order repaired and checked by reading, not compiled, not run on hardware
//
#include <hip/hip_runtime.h>

// MHA: B=2, T=4096, D=512, H=8, DH=64 — all matmuls via v_wmma_f32_16x16x32_bf16,
// attention fused (flash-style, online softmax) so scores (1 GB) never hit HBM.

#define B_  2
#define T_  4096
#define D_  512
#define H_  8
#define DH_ 64

typedef __bf16 bf16_t;
typedef __attribute__((ext_vector_type(16))) __bf16 v16bf;
typedef __attribute__((ext_vector_type(8)))  __bf16 v8bf;
typedef __attribute__((ext_vector_type(8)))  float  v8f;

union ABFrag { v16bf v; v8bf h[2]; };

static __device__ __forceinline__ v8f wmma_bf16(const ABFrag& a, const ABFrag& b, v8f c) {
  // (neg_a, A, neg_b, B, c_mod, C, reuse_a, reuse_b)
  return __builtin_amdgcn_wmma_f32_16x16x32_bf16(false, a.v, false, b.v, (short)0, c, false, false);
}

// A-fragment (16x32 bf16, row-major, leading dim ld). ISA layout: lane group g
// holds row r = lane&15; elements 0..7 = cols 8g..8g+7, elements 8..15 = cols 16+8g..16+8g+7.
static __device__ __forceinline__ ABFrag load_a(const bf16_t* __restrict__ p, int ld, int r, int g) {
  ABFrag f;
  f.h[0] = *(const v8bf*)(p + r * ld + 8 * g);
  f.h[1] = *(const v8bf*)(p + r * ld + 16 + 8 * g);
  return f;
}

// B-fragment (32x16): element (k,n) = m[n*ldm + k] (N-major storage => contiguous k per lane).
// ISA layout: lane group g holds column n = lane&15; elements 0..7 = k 8g..8g+7, elements 8..15 = k 16+8g..16+8g+7.
static __device__ __forceinline__ ABFrag load_b(const bf16_t* __restrict__ m, int ldm, int n, int g) {
  ABFrag f;
  f.h[0] = *(const v8bf*)(m + n * ldm + 8 * g);
  f.h[1] = *(const v8bf*)(m + n * ldm + 16 + 8 * g);
  return f;
}

// ---------------- prep kernels ----------------
__global__ void cvt_f32_bf16(const float* __restrict__ x, bf16_t* __restrict__ y, int n) {
  int i = blockIdx.x * blockDim.x + threadIdx.x;
  if (i < n) y[i] = (bf16_t)x[i];
}

__global__ void transpose_w(const float* __restrict__ w, bf16_t* __restrict__ wt) {
  int i = blockIdx.x * blockDim.x + threadIdx.x;   // 0..D*D-1
  int nn = i >> 9, kk = i & (D_ - 1);
  wt[i] = (bf16_t)w[kk * D_ + nn];                 // WT[n][k] = W[k][n]
}

// ---------------- QKV projection (one wave = 16 rows x 64 cols) ----------------
__global__ __launch_bounds__(32) void gemm_qkv(
    const bf16_t* __restrict__ xb,
    const bf16_t* __restrict__ WqT, const bf16_t* __restrict__ WkT, const bf16_t* __restrict__ WvT,
    const float* __restrict__ bq, const float* __restrict__ bk, const float* __restrict__ bv,
    bf16_t* __restrict__ Qs, bf16_t* __restrict__ Kb, bf16_t* __restrict__ VT) {
  const int lane = threadIdx.x & 31;
  const int r = lane & 15, g = lane >> 4, n = lane & 15;
  const int row0 = blockIdx.x * 16;
  const int col0 = blockIdx.y * 64;
  const int mode = blockIdx.z;                       // 0=Q 1=K 2=V
  const bf16_t* WT  = (mode == 0) ? WqT : (mode == 1) ? WkT : WvT;
  const float* bias = (mode == 0) ? bq  : (mode == 1) ? bk  : bv;
  const bf16_t* Ap = xb + (size_t)row0 * D_;

  v8f acc[4] = {};
#pragma unroll
  for (int kk = 0; kk < D_ / 32; ++kk) {
    ABFrag a = load_a(Ap + kk * 32, D_, r, g);
#pragma unroll
    for (int nt = 0; nt < 4; ++nt) {
      ABFrag b = load_b(WT + (size_t)(col0 + nt * 16) * D_ + kk * 32, D_, n, g);
      acc[nt] = wmma_bf16(a, b, acc[nt]);
    }
  }
#pragma unroll
  for (int nt = 0; nt < 4; ++nt) {
    int col = col0 + nt * 16 + n;
    float bval = bias[col];
    int hh = col >> 6, dh = col & 63;
#pragma unroll
    for (int j = 0; j < 8; ++j) {
      int tok = row0 + j + 8 * g;
      int bb = tok >> 12, t = tok & (T_ - 1);
      float v = acc[nt][j] + bval;
      size_t bh = (size_t)bb * H_ + hh;
      if (mode == 0)      Qs[(bh * T_ + t) * DH_ + dh] = (bf16_t)(v * 0.125f); // fold 1/sqrt(DH)
      else if (mode == 1) Kb[(bh * T_ + t) * DH_ + dh] = (bf16_t)v;
      else                VT[(bh * DH_ + dh) * T_ + t] = (bf16_t)v;            // V transposed
    }
  }
}

// ---------------- fused attention: one wave = 16 query rows, full DH ----------------
__global__ __launch_bounds__(32) void flash_attn(
    const bf16_t* __restrict__ Qs, const bf16_t* __restrict__ Kb, const bf16_t* __restrict__ VT,
    const unsigned char* __restrict__ mask, bf16_t* __restrict__ Ob) {
  __shared__ bf16_t lds[16 * 32];                    // P tile staging (C-layout -> A-layout)
  const int lane = threadIdx.x & 31;
  const int r = lane & 15, g = lane >> 4, n = lane & 15;
  const int qt = blockIdx.x & (T_ / 16 - 1);         // 256 q-tiles per (b,h)
  const int bh = blockIdx.x >> 8;
  const int hh = bh & (H_ - 1), bb = bh >> 3;

  const bf16_t* Qp = Qs + ((size_t)bh * T_ + qt * 16) * DH_;
  const bf16_t* Kp = Kb + (size_t)bh * T_ * DH_;
  const bf16_t* Vp = VT + (size_t)bh * DH_ * T_;

  ABFrag qf[2];
#pragma unroll
  for (int kc = 0; kc < 2; ++kc) qf[kc] = load_a(Qp + kc * 32, DH_, r, g);

  v8f o[4] = {};
  float m[8], ssum[8];
#pragma unroll
  for (int j = 0; j < 8; ++j) { m[j] = -3.0e38f; ssum[j] = 0.f; }

  for (int kt = 0; kt < T_ / 32; ++kt) {
    const int kv0 = kt * 32;
    v8f s0 = {}, s1 = {};
#pragma unroll
    for (int kc = 0; kc < 2; ++kc) {
      ABFrag k0 = load_b(Kp + (size_t)kv0 * DH_ + kc * 32, DH_, n, g);
      ABFrag k1 = load_b(Kp + (size_t)(kv0 + 16) * DH_ + kc * 32, DH_, n, g);
      s0 = wmma_bf16(qf[kc], k0, s0);
      s1 = wmma_bf16(qf[kc], k1, s1);
    }
    float p0[8], p1[8];
#pragma unroll
    for (int j = 0; j < 8; ++j) {
      int qrow = qt * 16 + j + 8 * g;
      const unsigned char* mr = mask + (size_t)qrow * T_ + kv0;
      float v0 = s0[j] + (mr[n]      ? 0.f : -1e9f);
      float v1 = s1[j] + (mr[16 + n] ? 0.f : -1e9f);
      float tmax = fmaxf(v0, v1);
#pragma unroll
      for (int d = 8; d >= 1; d >>= 1) tmax = fmaxf(tmax, __shfl_xor(tmax, d, 32));
      float mn = fmaxf(m[j], tmax);
      float alpha = __expf(m[j] - mn);
      float e0 = __expf(v0 - mn), e1 = __expf(v1 - mn);
      float ps = e0 + e1;
#pragma unroll
      for (int d = 8; d >= 1; d >>= 1) ps += __shfl_xor(ps, d, 32);
      ssum[j] = ssum[j] * alpha + ps;
      m[j] = mn;
      p0[j] = e0; p1[j] = e1;
#pragma unroll
      for (int dt = 0; dt < 4; ++dt) o[dt][j] *= alpha;
    }
    __syncthreads();
#pragma unroll
    for (int j = 0; j < 8; ++j) {
      lds[(j + 8 * g) * 32 + n]      = (bf16_t)p0[j];
      lds[(j + 8 * g) * 32 + 16 + n] = (bf16_t)p1[j];
    }
    __syncthreads();
    ABFrag pf;
    pf.h[0] = *(const v8bf*)(lds + r * 32 + 8 * g);
    pf.h[1] = *(const v8bf*)(lds + r * 32 + 16 + 8 * g);
#pragma unroll
    for (int dt = 0; dt < 4; ++dt) {
      ABFrag vf = load_b(Vp + (size_t)(dt * 16) * T_ + kv0, T_, n, g);
      o[dt] = wmma_bf16(pf, vf, o[dt]);
    }
  }
#pragma unroll
  for (int dt = 0; dt < 4; ++dt) {
    int col = hh * DH_ + dt * 16 + n;
#pragma unroll
    for (int j = 0; j < 8; ++j) {
      int t = qt * 16 + j + 8 * g;
      Ob[((size_t)bb * T_ + t) * D_ + col] = (bf16_t)(o[dt][j] / ssum[j]);
    }
  }
}

// ---------------- output projection ----------------
__global__ __launch_bounds__(32) void gemm_out(
    const bf16_t* __restrict__ Ob, const bf16_t* __restrict__ WoT,
    const float* __restrict__ bo, float* __restrict__ out) {
  const int lane = threadIdx.x & 31;
  const int r = lane & 15, g = lane >> 4, n = lane & 15;
  const int row0 = blockIdx.x * 16;
  const int col0 = blockIdx.y * 64;
  const bf16_t* Ap = Ob + (size_t)row0 * D_;

  v8f acc[4] = {};
#pragma unroll
  for (int kk = 0; kk < D_ / 32; ++kk) {
    ABFrag a = load_a(Ap + kk * 32, D_, r, g);
#pragma unroll
    for (int nt = 0; nt < 4; ++nt) {
      ABFrag b = load_b(WoT + (size_t)(col0 + nt * 16) * D_ + kk * 32, D_, n, g);
      acc[nt] = wmma_bf16(a, b, acc[nt]);
    }
  }
#pragma unroll
  for (int nt = 0; nt < 4; ++nt) {
    int col = col0 + nt * 16 + n;
    float bval = bo[col];
#pragma unroll
    for (int j = 0; j < 8; ++j) {
      int tok = row0 + j + 8 * g;
      out[(size_t)tok * D_ + col] = acc[nt][j] + bval;
    }
  }
}

extern "C" void kernel_launch(void* const* d_in, const int* in_sizes, int n_in,
                              void* d_out, int out_size, void* d_ws, size_t ws_size,
                              hipStream_t stream) {
  const float* x  = (const float*)d_in[0];
  const float* Wq = (const float*)d_in[1];
  const float* bq = (const float*)d_in[2];
  const float* Wk = (const float*)d_in[3];
  const float* bk = (const float*)d_in[4];
  const float* Wv = (const float*)d_in[5];
  const float* bv = (const float*)d_in[6];
  const float* Wo = (const float*)d_in[7];
  const float* bo = (const float*)d_in[8];
  const unsigned char* mask = (const unsigned char*)d_in[9]; // jax bool = 1 byte
  float* out = (float*)d_out;

  // workspace carve-up (~44 MB)
  char* ws = (char*)d_ws;
  size_t off = 0;
  auto carve = [&](size_t bytes) {
    void* p = ws + off;
    off += (bytes + 255) & ~(size_t)255;
    return p;
  };
  const size_t NTOK = (size_t)B_ * T_;               // 8192
  bf16_t* xb  = (bf16_t*)carve(NTOK * D_ * 2);
  bf16_t* WqT = (bf16_t*)carve((size_t)D_ * D_ * 2);
  bf16_t* WkT = (bf16_t*)carve((size_t)D_ * D_ * 2);
  bf16_t* WvT = (bf16_t*)carve((size_t)D_ * D_ * 2);
  bf16_t* WoT = (bf16_t*)carve((size_t)D_ * D_ * 2);
  bf16_t* Qs  = (bf16_t*)carve(NTOK * D_ * 2);       // (B,H,T,DH), pre-scaled
  bf16_t* Kb  = (bf16_t*)carve(NTOK * D_ * 2);       // (B,H,T,DH)
  bf16_t* VTb = (bf16_t*)carve(NTOK * D_ * 2);       // (B,H,DH,T) transposed
  bf16_t* Ob  = (bf16_t*)carve(NTOK * D_ * 2);       // (B,T,D) concat heads

  const int NX = (int)(NTOK * D_);
  cvt_f32_bf16<<<(NX + 255) / 256, 256, 0, stream>>>(x, xb, NX);
  transpose_w<<<(D_ * D_) / 256, 256, 0, stream>>>(Wq, WqT);
  transpose_w<<<(D_ * D_) / 256, 256, 0, stream>>>(Wk, WkT);
  transpose_w<<<(D_ * D_) / 256, 256, 0, stream>>>(Wv, WvT);
  transpose_w<<<(D_ * D_) / 256, 256, 0, stream>>>(Wo, WoT);

  gemm_qkv<<<dim3(NTOK / 16, D_ / 64, 3), 32, 0, stream>>>(
      xb, WqT, WkT, WvT, bq, bk, bv, Qs, Kb, VTb);

  flash_attn<<<B_ * H_ * (T_ / 16), 32, 0, stream>>>(Qs, Kb, VTb, mask, Ob);

  gemm_out<<<dim3(NTOK / 16, D_ / 64), 32, 0, stream>>>(Ob, WoT, bo, out);
}
